// B_Tier_35390530519893
// MI455X (gfx1250) — matrix operand order repaired and checked by reading, not compiled, not run on hardware
//
#include <hip/hip_runtime.h>

// ---------------------------------------------------------------------------
// Types for WMMA fragments
// ---------------------------------------------------------------------------
typedef __attribute__((ext_vector_type(16))) __bf16    v16bf;
typedef __attribute__((ext_vector_type(8)))  float     v8f;
typedef __attribute__((ext_vector_type(4)))  unsigned  u32x4;

union BF16x16 {
    v16bf          v;
    unsigned short s[16];
    u32x4          q[2];
};
union BF16x8 {
    u32x4          q;
    unsigned short s[8];
};

__device__ __forceinline__ unsigned short f2bf(float f) {
    unsigned u = __float_as_uint(f);
    u += 0x7fffu + ((u >> 16) & 1u);   // round-to-nearest-even
    return (unsigned short)(u >> 16);
}

// A fragment (16x32 bf16): lane m=lane&15, half=lane>>4.
// elems 0-7 = K[half*8..], elems 8-15 = K[16+half*8..]
__device__ __forceinline__ void load_frag_a(BF16x16& r, const unsigned short* src,
                                            int ld, int lane) {
    const int m = lane & 15, half = lane >> 4;
    r.q[0] = *(const u32x4*)(src + m * ld + half * 8);
    r.q[1] = *(const u32x4*)(src + m * ld + 16 + half * 8);
}

// B fragment (32x16 bf16) = W^T tile; src -> W[n0][k0] row-major.
// lane holds col n=lane&15; elems 0-7 = W[n][half*8..], elems 8-15 = W[n][16+half*8..]
__device__ __forceinline__ void load_frag_b(BF16x16& r, const unsigned short* src,
                                            int ld, int lane) {
    const int n = lane & 15, half = lane >> 4;
    r.q[0] = *(const u32x4*)(src + n * ld + half * 8);
    r.q[1] = *(const u32x4*)(src + n * ld + 16 + half * 8);
}

__device__ __forceinline__ v8f wmma_bf16(v16bf a, v16bf b, v8f c) {
    return __builtin_amdgcn_wmma_f32_16x16x32_bf16(
        false, a, false, b, (short)0, c, false, false);
}

// ---------------------------------------------------------------------------
// Problem constants
// ---------------------------------------------------------------------------
#define I_DIM   256
#define H_DIM   512
#define BD      256      // output dim
#define BATCH   64
#define SEQ     2048
#define IN_DIM  1280     // 3*I_DIM + H_DIM
#define XK      768      // x-part of K
#define NG      1536     // 3*H_DIM (z,r,h stacked)
#define MTOT    (BATCH * SEQ)   // 131072

// ---------------------------------------------------------------------------
// Kernel 1a: weights -> bf16 (Wz/Wr/Wh stacked rows 0..1535; Wb), biases -> bcat
// ---------------------------------------------------------------------------
__global__ void prep_w_kernel(const float* __restrict__ Wz, const float* __restrict__ Wr,
                              const float* __restrict__ Wh, const float* __restrict__ bz,
                              const float* __restrict__ br, const float* __restrict__ bh,
                              const float* __restrict__ Wb,
                              unsigned short* __restrict__ Wbf,
                              unsigned short* __restrict__ Wbbf,
                              float* __restrict__ bcat) {
    const int NW = NG * IN_DIM;
    const int NB = BD * H_DIM;
    int stride = gridDim.x * blockDim.x;
    for (int i = blockIdx.x * blockDim.x + threadIdx.x; i < NW + NB + NG; i += stride) {
        if (i < NW) {
            int row = i / IN_DIM, k = i % IN_DIM;
            int g = row >> 9, n = row & 511;
            const float* W = (g == 0) ? Wz : (g == 1) ? Wr : Wh;
            Wbf[i] = f2bf(W[n * IN_DIM + k]);
        } else if (i < NW + NB) {
            Wbbf[i - NW] = f2bf(Wb[i - NW]);
        } else {
            int j = i - NW - NB;
            const float* bsrc = (j < 512) ? bz : (j < 1024) ? br : bh;
            bcat[j] = bsrc[j & 511];
        }
    }
}

// ---------------------------------------------------------------------------
// Kernel 1b: x -> Xbf[m][768] bf16, m = t*64+b,  Xbf[m][k] = x[b, k/256, t, k%256]
// Processed in 8-element groups (never crosses a 256-col segment).
// ---------------------------------------------------------------------------
__global__ void prep_x_kernel(const float* __restrict__ x,
                              unsigned short* __restrict__ Xbf) {
    const int NGRP = MTOT * (XK / 8);          // 12,582,912 groups of 8
    int stride = gridDim.x * blockDim.x;
    for (int g = blockIdx.x * blockDim.x + threadIdx.x; g < NGRP; g += stride) {
        const int m  = g / (XK / 8);
        const int kc = (g % (XK / 8)) * 8;
        const int b = m & 63, t = m >> 6;
        const int s = kc >> 8, kk = kc & 255;
        const float* xp = x + (((size_t)b * 3 + s) * SEQ + t) * I_DIM + kk;
        BF16x8 o;
#pragma unroll
        for (int j = 0; j < 8; ++j) o.s[j] = f2bf(xp[j]);
        *(u32x4*)(Xbf + (size_t)m * XK + kc) = o.q;
    }
}

// ---------------------------------------------------------------------------
// Kernel 2: Gx[m][gn] = Xbf[m][:] . Wbf[gn][0:768] + bcat[gn]
// GEMM M=131072, N=1536, K=768. Block=256 thr (8 waves), tile 64x256.
// Per k-step: batch-load A + 8 B frags, then 8 WMMAs (one wait per k-step).
// ---------------------------------------------------------------------------
__global__ __launch_bounds__(256) void gemm_x_kernel(
    const unsigned short* __restrict__ Xbf,   // 131072 x 768 bf16
    const unsigned short* __restrict__ Wbf,   // 1536 x 1280 bf16
    const float* __restrict__ bcat,           // 1536
    float* __restrict__ Gx) {                 // 131072 x 1536 fp32
    const int lane = threadIdx.x & 31;
    const int w    = threadIdx.x >> 5;
    const int half = lane >> 4;
    const int m0    = blockIdx.x * 64 + (w & 3) * 16;
    const int nbase = blockIdx.y * 256 + (w >> 2) * 128;

    v8f acc[8] = {};

    for (int kt = 0; kt < XK / 32; ++kt) {
        const int k0 = kt * 32;
        BF16x16 af;
        load_frag_a(af, Xbf + (size_t)m0 * XK + k0, XK, lane);
        BF16x16 bf[8];
#pragma unroll
        for (int j = 0; j < 8; ++j)
            load_frag_b(bf[j], Wbf + (size_t)(nbase + j * 16) * IN_DIM + k0, IN_DIM, lane);
#pragma unroll
        for (int j = 0; j < 8; ++j)
            acc[j] = wmma_bf16(af.v, bf[j].v, acc[j]);
    }
#pragma unroll
    for (int j = 0; j < 8; ++j) {
        const int n = nbase + j * 16 + (lane & 15);
        const float bias = bcat[n];
#pragma unroll
        for (int v = 0; v < 8; ++v) {
            const int m = m0 + v + half * 8;
            Gx[(size_t)m * NG + n] = acc[j][v] + bias;
        }
    }
}

// ---------------------------------------------------------------------------
// Kernel 3: sequential scan (critical path). One workgroup, 1024 thr = 32 waves.
// h fp32 lives in REGISTERS (each element owned by exactly one wave).
// LDS: bf16 h mirror (64K) + bf16 r*h (64K) = 128K.  2 barriers / step.
// Per step/wave: 128 WMMA (z,r) + 64 WMMA (h~). Output projection deferred.
// Emits Hall[t][b][h] bf16 for the parallel out-projection GEMM.
// ---------------------------------------------------------------------------
__global__ __launch_bounds__(1024) void scan_kernel(
    const unsigned short* __restrict__ Wbf,   // 1536 x 1280 (h-part cols 768..1279)
    const float* __restrict__ Gx,             // 131072 x 1536
    unsigned short* __restrict__ Hall,        // 2048 x 64 x 512 bf16
    float* __restrict__ hfin) {               // 64 x 512
    __shared__ unsigned short hbf[BATCH * H_DIM];
    __shared__ unsigned short rhbf[BATCH * H_DIM];

    const int tid  = threadIdx.x;
    const int lane = tid & 31;
    const int w    = tid >> 5;
    const int half = lane >> 4;
    const int ln   = lane & 15;
    const int m0   = (w & 3) * 16;
    const int cg   = w >> 2;

    float hreg[4][8] = {};   // this wave's fp32 h tile: (m0+v+8*half, cg*64+j*16+ln)

    for (int i = tid; i < BATCH * H_DIM; i += 1024) hbf[i] = 0;
    __syncthreads();

    for (int t = 0; t < SEQ; ++t) {
        const float* gx = Gx + (size_t)t * (BATCH * NG);

        // ---- prefetch next step's Gx slab (this wave's region) ----
        if (t + 1 < SEQ) {
            const float* gxn = gx + BATCH * NG;
#pragma unroll
            for (int g = 0; g < 3; ++g)
                __builtin_prefetch(gxn + (m0 + ln) * NG + g * 512 + cg * 64 + half * 32, 0, 0);
        }

        // ---- z and r: accum init from precomputed x-part (incl. bias) ----
        v8f zc[4], rc[4];
#pragma unroll
        for (int j = 0; j < 4; ++j) {
            const int n = cg * 64 + j * 16 + ln;
#pragma unroll
            for (int v = 0; v < 8; ++v) {
                const int m = m0 + v + half * 8;
                zc[j][v] = gx[m * NG + n];
                rc[j][v] = gx[m * NG + 512 + n];
            }
        }
        for (int kt = 0; kt < 16; ++kt) {
            const int k = XK + kt * 32;
            BF16x16 a;
            load_frag_a(a, hbf + m0 * H_DIM + kt * 32, H_DIM, lane);
            BF16x16 bzf[4], brf[4];
#pragma unroll
            for (int j = 0; j < 4; ++j) {
                const int n0 = cg * 64 + j * 16;
                load_frag_b(bzf[j], Wbf + (size_t)n0 * IN_DIM + k, IN_DIM, lane);
                load_frag_b(brf[j], Wbf + (size_t)(512 + n0) * IN_DIM + k, IN_DIM, lane);
            }
#pragma unroll
            for (int j = 0; j < 4; ++j) {
                zc[j] = wmma_bf16(a.v, bzf[j].v, zc[j]);
                rc[j] = wmma_bf16(a.v, brf[j].v, rc[j]);
            }
        }
        // ---- sigmoid; write r*h (bf16) for the h~ GEMM (h from registers) ----
#pragma unroll
        for (int j = 0; j < 4; ++j) {
            const int n = cg * 64 + j * 16 + ln;
#pragma unroll
            for (int v = 0; v < 8; ++v) {
                const int m = m0 + v + half * 8;
                const float z = 1.f / (1.f + __expf(-zc[j][v]));
                const float r = 1.f / (1.f + __expf(-rc[j][v]));
                zc[j][v] = z;
                rhbf[m * H_DIM + n] = f2bf(r * hreg[j][v]);
            }
        }
        __syncthreads();   // barrier A: rhbf ready; all hbf reads of step t done

        // ---- h~ GEMM over r*h ----
        v8f hc[4];
#pragma unroll
        for (int j = 0; j < 4; ++j) {
            const int n = cg * 64 + j * 16 + ln;
#pragma unroll
            for (int v = 0; v < 8; ++v)
                hc[j][v] = gx[(m0 + v + half * 8) * NG + 1024 + n];
        }
        for (int kt = 0; kt < 16; ++kt) {
            const int k = XK + kt * 32;
            BF16x16 a;
            load_frag_a(a, rhbf + m0 * H_DIM + kt * 32, H_DIM, lane);
            BF16x16 bhf[4];
#pragma unroll
            for (int j = 0; j < 4; ++j)
                load_frag_b(bhf[j], Wbf + (size_t)(1024 + cg * 64 + j * 16) * IN_DIM + k,
                            IN_DIM, lane);
#pragma unroll
            for (int j = 0; j < 4; ++j)
                hc[j] = wmma_bf16(a.v, bhf[j].v, hc[j]);
        }
        // ---- h_new = (1-z)*h + z*tanh(h~); publish bf16 h ----
        unsigned short* hall_t = Hall + (size_t)t * (BATCH * H_DIM);
#pragma unroll
        for (int j = 0; j < 4; ++j) {
            const int n = cg * 64 + j * 16 + ln;
#pragma unroll
            for (int v = 0; v < 8; ++v) {
                const int m = m0 + v + half * 8;
                const float ht = tanhf(hc[j][v]);
                const float z  = zc[j][v];
                const float hn = (1.f - z) * hreg[j][v] + z * ht;
                hreg[j][v] = hn;
                const unsigned short hb = f2bf(hn);
                hbf[m * H_DIM + n] = hb;          // safe: all readers passed barrier A
                hall_t[m * H_DIM + n] = hb;       // async store, drains in background
            }
        }
        __syncthreads();   // barrier B: new hbf visible for step t+1
    }

    // final hidden state from registers
#pragma unroll
    for (int j = 0; j < 4; ++j) {
        const int n = cg * 64 + j * 16 + ln;
#pragma unroll
        for (int v = 0; v < 8; ++v)
            hfin[(m0 + v + half * 8) * H_DIM + n] = hreg[j][v];
    }
}

// ---------------------------------------------------------------------------
// Kernel 4: outs[b][t][:] = Hall[t][b][:] @ Wb.T + bb   (parallel, whole chip)
// GEMM M=131072 (m=t*64+b), N=256, K=512. Block=256 thr, tile 64x256 ->
// one block per t; b = local row.
// ---------------------------------------------------------------------------
__global__ __launch_bounds__(256) void gemm_out_kernel(
    const unsigned short* __restrict__ Hall,  // 131072 x 512 bf16
    const unsigned short* __restrict__ Wbbf,  // 256 x 512 bf16
    const float* __restrict__ bb,             // 256
    float* __restrict__ outs) {               // 64 x 2048 x 256
    const int lane = threadIdx.x & 31;
    const int w    = threadIdx.x >> 5;
    const int half = lane >> 4;
    const int t    = blockIdx.x;
    const int m0   = t * 64 + (w & 3) * 16;
    const int nbase = (w >> 2) * 128;

    v8f acc[8];
#pragma unroll
    for (int j = 0; j < 8; ++j) {
        const float bias = bb[nbase + j * 16 + (lane & 15)];
#pragma unroll
        for (int v = 0; v < 8; ++v) acc[j][v] = bias;
    }

    for (int kt = 0; kt < 16; ++kt) {
        const int k0 = kt * 32;
        BF16x16 af;
        load_frag_a(af, Hall + (size_t)m0 * H_DIM + k0, H_DIM, lane);
        BF16x16 bf[8];
#pragma unroll
        for (int j = 0; j < 8; ++j)
            load_frag_b(bf[j], Wbbf + (size_t)(nbase + j * 16) * H_DIM + k0, H_DIM, lane);
#pragma unroll
        for (int j = 0; j < 8; ++j)
            acc[j] = wmma_bf16(af.v, bf[j].v, acc[j]);
    }
#pragma unroll
    for (int j = 0; j < 8; ++j) {
        const int n = nbase + j * 16 + (lane & 15);
#pragma unroll
        for (int v = 0; v < 8; ++v) {
            const int b = (w & 3) * 16 + v + half * 8;   // local row = batch index
            outs[((size_t)b * SEQ + t) * BD + n] = acc[j][v];
        }
    }
}

// ---------------------------------------------------------------------------
// Launch
// ---------------------------------------------------------------------------
extern "C" void kernel_launch(void* const* d_in, const int* in_sizes, int n_in,
                              void* d_out, int out_size, void* d_ws, size_t ws_size,
                              hipStream_t stream) {
    const float* x  = (const float*)d_in[0];
    const float* Wz = (const float*)d_in[1];
    const float* bz = (const float*)d_in[2];
    const float* Wr = (const float*)d_in[3];
    const float* br = (const float*)d_in[4];
    const float* Wh = (const float*)d_in[5];
    const float* bh = (const float*)d_in[6];
    const float* Wb = (const float*)d_in[7];
    const float* bb = (const float*)d_in[8];

    float* outs = (float*)d_out;                          // 64*2048*256
    float* hfin = outs + (size_t)MTOT * BD;               // 64*512

    // Workspace layout (all offsets 256B-aligned)
    char* ws = (char*)d_ws;
    unsigned short* Wbf  = (unsigned short*)ws;                        // 3,932,160 B
    unsigned short* Wbbf = Wbf + (size_t)NG * IN_DIM;                  //   262,144 B
    float*          bcat = (float*)(Wbbf + (size_t)BD * H_DIM);        //     6,144 B
    unsigned short* Xbf  = (unsigned short*)((char*)bcat + 6400 * sizeof(float) / sizeof(float) * 4);
    Xbf = (unsigned short*)((char*)ws + 4200448);                      // 201,326,592 B
    float*          Gx   = (float*)((char*)ws + 4200448 + 201326592);  // 805,306,368 B
    unsigned short* Hall = (unsigned short*)((char*)Gx + (size_t)MTOT * NG * 4); // 134,217,728 B

    // 1a) weight / bias conversion
    {
        const int total = NG * IN_DIM + BD * H_DIM + NG;
        prep_w_kernel<<<(total + 255) / 256, 256, 0, stream>>>(
            Wz, Wr, Wh, bz, br, bh, Wb, Wbf, Wbbf, bcat);
    }
    // 1b) x -> bf16, scan-friendly layout
    prep_x_kernel<<<49152, 256, 0, stream>>>(x, Xbf);

    // 2) parallel x-contribution GEMM: (131072 x 768) @ (768 x 1536) + bias
    {
        dim3 grid(MTOT / 64, NG / 256);
        gemm_x_kernel<<<grid, 256, 0, stream>>>(Xbf, Wbf, bcat, Gx);
    }
    // 3) sequential recurrence on one WGP (critical path)
    scan_kernel<<<1, 1024, 0, stream>>>(Wbf, Gx, Hall, hfin);

    // 4) parallel output projection
    gemm_out_kernel<<<SEQ, 256, 0, stream>>>(Hall, Wbbf, bb, outs);
}
